// DTNNStep_12721693131112
// MI455X (gfx1250) — matrix operand order repaired and checked by reading, not compiled, not run on hardware
//
#include <hip/hip_runtime.h>
#include <hip/hip_bf16.h>
#include <math.h>

// Problem constants (from reference): B=16, N=128, NF=64, ND=100, NH=64
#define NATOM 128
#define NFEAT 64
#define NDIST 100
#define NHID  64
#define KPAD  128   // ND padded to multiple of 32 for f16 WMMA K-loop

typedef __attribute__((ext_vector_type(16))) _Float16 v16h;
typedef __attribute__((ext_vector_type(8)))  _Float16 v8h;
typedef __attribute__((ext_vector_type(8)))  float    v8f;

__device__ __forceinline__ v16h cat8(v8h lo, v8h hi) {
    v16h r;
#pragma unroll
    for (int i = 0; i < 8; ++i) { r[i] = lo[i]; r[i + 8] = hi[i]; }
    return r;
}

// A-matrix fragment, f16 16x32 (MxK), from row-major LDS [rows][ld].
// ISA layout: lane L holds M = L%16; halves 0..7 -> K = (L/16)*8 + 0..7,
// halves 8..15 -> K = 16 + (L/16)*8 + 0..7. Both runs contiguous -> 2x b128.
__device__ __forceinline__ v16h load_a_frag(const _Float16* s, int ld,
                                            int mbase, int kbase, int lane) {
    const _Float16* p = s + (mbase + (lane & 15)) * ld + kbase + ((lane >> 4) << 3);
    v8h lo = *(const v8h*)(p);
    v8h hi = *(const v8h*)(p + 16);
    return cat8(lo, hi);
}

// B-matrix fragment, f16 32x16 (KxN), from LDS stored TRANSPOSED [ncols][ldk].
// ISA layout: lane L holds N = L%16; halves 0..7 -> K = (L/16)*8 + 0..7,
// halves 8..15 -> K = 16 + (L/16)*8 + 0..7 (matches A fragment).
__device__ __forceinline__ v16h load_b_fragT(const _Float16* sT, int ldk,
                                             int kbase, int nbase, int lane) {
    const _Float16* p = sT + (nbase + (lane & 15)) * ldk + kbase + ((lane >> 4) << 3);
    v8h lo = *(const v8h*)(p);
    v8h hi = *(const v8h*)(p + 16);
    return cat8(lo, hi);
}

__device__ __forceinline__ v8f wmma16(v16h a, v16h b, v8f c) {
    return __builtin_amdgcn_wmma_f32_16x16x32_f16(false, a, false, b,
                                                  (short)0, c, false, false);
}

// One workgroup (256 threads = 8 wave32) per (b, i) pair.
__global__ __launch_bounds__(256)
void dtnn_step_kernel(const float* __restrict__ AF,    // [B,N,NF]
                      const float* __restrict__ DM,    // [B,N,N,ND]
                      const float* __restrict__ MASK,  // [B,N,N]
                      const float* __restrict__ Wcf,   // [NF,NH]
                      const float* __restrict__ Wdf,   // [ND,NH]
                      const float* __restrict__ Wfc,   // [NH,NF]
                      const float* __restrict__ bcf,   // [NH]
                      const float* __restrict__ bdf,   // [NH]
                      float* __restrict__ OUT)         // [B,N,NF]
{
    __shared__ __align__(16) _Float16 s_A[NATOM * KPAD];     // dist[b,i] f16, K-padded (32KB)
    __shared__ __align__(16) _Float16 s_WdfT[NHID * KPAD];   // W_df^T [h][k]     (16KB)
    __shared__ __align__(16) _Float16 s_WcfT[NHID * NFEAT];  // W_cf^T [h][f]     (8KB)
    __shared__ __align__(16) _Float16 s_WfcT[NFEAT * NHID];  // W_fc^T [f][h]     (8KB)
    __shared__ __align__(16) _Float16 s_Aat[NATOM * NFEAT];  // atom_features[b]  (16KB)
    __shared__ __align__(16) _Float16 s_G[NATOM * NHID];     // gated f16         (16KB)
    __shared__ float s_red[NFEAT];                           // j-reduction

    const int tid = threadIdx.x;
    const int bi  = blockIdx.x;          // b*N + i
    const int b   = bi >> 7;

    // ---- Stage fp32 -> f16 into LDS (coalesced global reads) ----
    const float* dmrow = DM + (size_t)bi * (NATOM * NDIST);
#pragma unroll 4
    for (int idx = tid; idx < NATOM * KPAD; idx += 256) {
        int j = idx >> 7, d = idx & 127;
        float v = (d < NDIST) ? dmrow[j * NDIST + d] : 0.0f;   // zero K-pad
        s_A[idx] = (_Float16)v;
    }
    for (int idx = tid; idx < KPAD * NHID; idx += 256) {
        int k = idx >> 6, h = idx & 63;
        float v = (idx < NDIST * NHID) ? Wdf[idx] : 0.0f;      // zero K-pad
        s_WdfT[h * KPAD + k] = (_Float16)v;
    }
    for (int idx = tid; idx < NFEAT * NHID; idx += 256) {
        int r = idx >> 6, c = idx & 63;                        // row=K, col=N in both
        s_WcfT[c * NFEAT + r] = (_Float16)Wcf[idx];
        s_WfcT[c * NHID + r]  = (_Float16)Wfc[idx];
    }
    const float* afb = AF + (size_t)b * (NATOM * NFEAT);
    for (int idx = tid; idx < NATOM * NFEAT; idx += 256)
        s_Aat[idx] = (_Float16)afb[idx];
    if (tid < NFEAT) s_red[tid] = 0.0f;
    __syncthreads();

    const int lane  = tid & 31;
    const int w     = tid >> 5;           // wave id -> row tile (j = 16w .. 16w+15)
    const int mrow  = w * 16;
    const int hoff  = lane & 15;          // N-column within tile
    const int rbase = (lane >> 4) << 3;   // C-layout: lanes>=16 hold M = r+8

    // ---- GEMM1 (dist x W_df) fused with atom_h gating -> s_G (f16) ----
#pragma unroll
    for (int nt = 0; nt < 4; ++nt) {
        const int h = nt * 16 + hoff;

        v8f ah = {};                       // atom_h tile (same C layout as acc)
#pragma unroll
        for (int kt = 0; kt < 2; ++kt) {
            v16h a  = load_a_frag(s_Aat, NFEAT, mrow, kt * 32, lane);
            v16h bm = load_b_fragT(s_WcfT, NFEAT, kt * 32, nt * 16, lane);
            ah = wmma16(a, bm, ah);
        }

        v8f acc = {};                      // dist_h tile
#pragma unroll
        for (int kt = 0; kt < 4; ++kt) {
            v16h a  = load_a_frag(s_A, KPAD, mrow, kt * 32, lane);
            v16h bm = load_b_fragT(s_WdfT, KPAD, kt * 32, nt * 16, lane);
            acc = wmma16(a, bm, acc);
        }

        const float vbdf = bdf[h];
        const float vbcf = bcf[h];
#pragma unroll
        for (int r = 0; r < 8; ++r) {
            int j = mrow + rbase + r;
            float g = (acc[r] + vbdf) * (ah[r] + vbcf);
            s_G[j * NHID + h] = (_Float16)g;   // same-wave producer/consumer
        }
    }

    // ---- GEMM2 (gated x W_fc), mask, tanh, reduce over j ----
    const float* mp = MASK + (size_t)bi * NATOM + mrow + rbase;
    float mv[8];
#pragma unroll
    for (int r = 0; r < 8; ++r) mv[r] = mp[r];

#pragma unroll
    for (int nt = 0; nt < 4; ++nt) {
        v8f acc = {};
#pragma unroll
        for (int kt = 0; kt < 2; ++kt) {
            v16h a  = load_a_frag(s_G, NHID, mrow, kt * 32, lane);
            v16h bm = load_b_fragT(s_WfcT, NHID, kt * 32, nt * 16, lane);
            acc = wmma16(a, bm, acc);
        }
        float psum = 0.0f;
#pragma unroll
        for (int r = 0; r < 8; ++r)
            psum += tanhf(acc[r] * mv[r]);
        atomicAdd(&s_red[nt * 16 + hoff], psum);   // ds_add_f32
    }
    __syncthreads();

    if (tid < NFEAT)
        OUT[(size_t)bi * NFEAT + tid] = s_red[tid] + AF[(size_t)bi * NFEAT + tid];
}

extern "C" void kernel_launch(void* const* d_in, const int* in_sizes, int n_in,
                              void* d_out, int out_size, void* d_ws, size_t ws_size,
                              hipStream_t stream) {
    (void)in_sizes; (void)n_in; (void)out_size; (void)d_ws; (void)ws_size;
    const float* AF   = (const float*)d_in[0];
    const float* DM   = (const float*)d_in[1];
    const float* MASK = (const float*)d_in[2];
    const float* Wcf  = (const float*)d_in[3];
    const float* Wdf  = (const float*)d_in[4];
    const float* Wfc  = (const float*)d_in[5];
    const float* bcf  = (const float*)d_in[6];
    const float* bdf  = (const float*)d_in[7];
    float* OUT = (float*)d_out;

    dim3 grid(16 * 128);   // one workgroup per (b, i)
    dim3 block(256);       // 8 wave32
    dtnn_step_kernel<<<grid, block, 0, stream>>>(AF, DM, MASK, Wcf, Wdf, Wfc,
                                                 bcf, bdf, OUT);
}
